// NETWORK_OF_DANS_batch_16338055594194
// MI455X (gfx1250) — matrix operand order repaired and checked by reading, not compiled, not run on hardware
//
#include <hip/hip_runtime.h>

// MI455X / gfx1250: wave32, WMMA f32<-f16 16x16x32 path.
typedef _Float16 v16h __attribute__((ext_vector_type(16)));
typedef _Float16 v8h  __attribute__((ext_vector_type(8)));
typedef _Float16 v4h  __attribute__((ext_vector_type(4)));
typedef float    v8f  __attribute__((ext_vector_type(8)));
typedef float    v4f  __attribute__((ext_vector_type(4)));

#define KDIM   1024     // reduction dim (all DIMS are 1024)
#define MROWS  512      // batch
#define NCOLS  32768    // dims[l+1] * C
#define M_TILE 128
#define N_TILE 128      // 4 DAN nodes of 32 channels
#define NODE_C 32

__device__ __forceinline__ float lrelu(float x) { return x > 0.0f ? x : 0.01f * x; }

// ---- tiny one-shot f32 -> f16 converter for x (512x1024) ----
__global__ __launch_bounds__(256, 1) void f32_to_f16_kernel(
    const float* __restrict__ in, _Float16* __restrict__ out) {
  const int i = (blockIdx.x * 256 + threadIdx.x) * 4;
  v4f v = *reinterpret_cast<const v4f*>(in + i);
  v4h h;
#pragma unroll
  for (int j = 0; j < 4; ++j) h[j] = (_Float16)v[j];
  *reinterpret_cast<v4h*>(out + i) = h;
}

// B-side: load 8 f32 at p0 and 8 f32 at p1, convert to 16 halves (W stays f32).
__device__ __forceinline__ v16h cvt2x8(const float* __restrict__ p0,
                                       const float* __restrict__ p1) {
  v4f x0 = *reinterpret_cast<const v4f*>(p0);
  v4f x1 = *reinterpret_cast<const v4f*>(p0 + 4);
  v4f x2 = *reinterpret_cast<const v4f*>(p1);
  v4f x3 = *reinterpret_cast<const v4f*>(p1 + 4);
  v16h r;
#pragma unroll
  for (int i = 0; i < 4; ++i) {
    r[i]      = (_Float16)x0[i];
    r[4 + i]  = (_Float16)x1[i];
    r[8 + i]  = (_Float16)x2[i];
    r[12 + i] = (_Float16)x3[i];
  }
  return r;
}

// A-side: pure f16 loads, no conversion. Two contiguous 16B chunks.
__device__ __forceinline__ v16h ldA(const _Float16* __restrict__ p0,
                                    const _Float16* __restrict__ p1) {
  v8h a0 = *reinterpret_cast<const v8h*>(p0);
  v8h a1 = *reinterpret_cast<const v8h*>(p1);
  return __builtin_shufflevector(a0, a1, 0, 1, 2, 3, 4, 5, 6, 7,
                                 8, 9, 10, 11, 12, 13, 14, 15);
}

// Fused: pre = A0*Wa^T + ba [+ A1*Wb^T + bb], then per-32-channel DAN MLP.
// OUT_F16: intermediate h layers stored f16 (consumed only as next-layer A).
template <bool OUT_F16>
__global__ __launch_bounds__(256, 1) void fused_vecs_dan_kernel(
    const _Float16* __restrict__ A0, const float* __restrict__ Wa, const float* __restrict__ ba,
    const _Float16* __restrict__ A1, const float* __restrict__ Wb, const float* __restrict__ bb,
    const float* __restrict__ Dw1, const float* __restrict__ Db1,
    const float* __restrict__ Dw2, const float* __restrict__ Db2,
    const float* __restrict__ Dw3, const float* __restrict__ Db3,
    int layer_code, void* __restrict__ hOutRaw) {
  // Half of M_TILE staged at a time: 64 x 132 f32 = 33.8 KB LDS.
  __shared__ float smem[64][N_TILE + 4];

  const int lane = threadIdx.x & 31;
  const int wave = threadIdx.x >> 5;
  const int l15  = lane & 15;          // A: M row / B: N col within 16-tile
  const int hi   = lane >> 4;          // lane-half selects K sub-pattern
  const int mBase = blockIdx.y * M_TILE;
  const int nCol  = blockIdx.x * N_TILE + wave * 16 + l15;  // this lane's W row

  v8f acc[8];
#pragma unroll
  for (int i = 0; i < 8; ++i)
#pragma unroll
    for (int r = 0; r < 8; ++r) acc[i][r] = 0.0f;

  // pass 0: main GEMM, pass 1: skip GEMM (accumulates into same tiles)
#pragma unroll 1
  for (int pass = 0; pass < 2; ++pass) {
    const _Float16* A = pass ? A1 : A0;
    const float*    W = pass ? Wb : Wa;
    if (A == nullptr) break;
    const float* wRow = W + (size_t)nCol * KDIM;   // W[n, :] contiguous in K
#pragma unroll 1
    for (int k0 = 0; k0 < KDIM; k0 += 32) {
      // B fragment (32x16): lane holds K = hi*16 .. hi*16+15 of column nCol
      v16h bf = cvt2x8(wRow + k0 + hi * 16, wRow + k0 + hi * 16 + 8);
#pragma unroll
      for (int mt = 0; mt < 8; ++mt) {
        // A fragment (16x32): lane holds K = hi*16 .. hi*16+15
        const _Float16* aRow = A + (size_t)(mBase + mt * 16 + l15) * KDIM;
        v16h af = ldA(aRow + k0 + hi * 16, aRow + k0 + hi * 16 + 8);
        acc[mt] = __builtin_amdgcn_wmma_f32_16x16x32_f16(
            false, af, false, bf, (short)0, acc[mt], false, false);
      }
    }
  }

  const float bsum = ba[nCol] + (A1 ? bb[nCol] : 0.0f);

#pragma unroll 1
  for (int half = 0; half < 2; ++half) {
    if (half) __syncthreads();  // previous half's DAN reads done
    // C/D layout: VGPR r -> M = r + 8*hi ; lane&15 -> N
#pragma unroll
    for (int mt = 0; mt < 4; ++mt) {
      const int mtg = half * 4 + mt;
#pragma unroll
      for (int r = 0; r < 8; ++r) {
        smem[mt * 16 + r + 8 * hi][wave * 16 + l15] = acc[mtg][r] + bsum;
      }
    }
    __syncthreads();

    // DAN MLP: 64 rows x 4 nodes = 256 tasks, one per thread.
    {
      const int row  = threadIdx.x >> 2;
      const int node = threadIdx.x & 3;
      const float* z = &smem[row][node * NODE_C];
      float h1v[15];
#pragma unroll 1
      for (int j = 0; j < 15; ++j) {
        float s = Db1[j] + Dw1[j * 35 + 32 + layer_code];  // one-hot layer code
#pragma unroll
        for (int c = 0; c < NODE_C; ++c) s += Dw1[j * 35 + c] * z[c];
        h1v[j] = lrelu(s);
      }
      float h2v[8];
#pragma unroll
      for (int i = 0; i < 8; ++i) {
        float s = Db2[i];
#pragma unroll
        for (int j = 0; j < 15; ++j) s += Dw2[i * 15 + j] * h1v[j];
        h2v[i] = lrelu(s);
      }
      float o = Db3[0];
#pragma unroll
      for (int i = 0; i < 8; ++i) o += Dw3[i] * h2v[i];
      o = lrelu(o);
      const size_t oidx =
          (size_t)(mBase + half * 64 + row) * 1024 + blockIdx.x * 4 + node;
      if (OUT_F16) {
        ((_Float16*)hOutRaw)[oidx] = (_Float16)o;
      } else {
        ((float*)hOutRaw)[oidx] = o;
      }
    }
  }
}

extern "C" void kernel_launch(void* const* d_in, const int* in_sizes, int n_in,
                              void* d_out, int out_size, void* d_ws, size_t ws_size,
                              hipStream_t stream) {
  const float* x    = (const float*)d_in[0];
  const float* W0   = (const float*)d_in[1];
  const float* b0   = (const float*)d_in[2];
  const float* W1   = (const float*)d_in[3];
  const float* b1   = (const float*)d_in[4];
  const float* W2   = (const float*)d_in[5];
  const float* b2   = (const float*)d_in[6];
  const float* Ws02 = (const float*)d_in[7];
  const float* bs02 = (const float*)d_in[8];
  const float* Ws13 = (const float*)d_in[9];
  const float* bs13 = (const float*)d_in[10];
  const float* Dw1  = (const float*)d_in[11];
  const float* Db1  = (const float*)d_in[12];
  const float* Dw2  = (const float*)d_in[13];
  const float* Db2  = (const float*)d_in[14];
  const float* Dw3  = (const float*)d_in[15];
  const float* Db3  = (const float*)d_in[16];

  // workspace: f16 copies of GEMM A inputs (x, h0, h1), 1 MB each
  _Float16* xh  = (_Float16*)d_ws;
  _Float16* h0h = xh + (size_t)MROWS * KDIM;
  _Float16* h1h = h0h + (size_t)MROWS * 1024;

  dim3 grid(NCOLS / N_TILE, MROWS / M_TILE);  // (256, 4)
  dim3 block(256);

  // x -> f16 once (512*1024 / (4*256) = 512 blocks)
  f32_to_f16_kernel<<<(MROWS * KDIM) / (4 * 256), 256, 0, stream>>>(x, xh);

  // Layer 0: pre0 = x*W0^T + b0
  fused_vecs_dan_kernel<true><<<grid, block, 0, stream>>>(
      xh, W0, b0, nullptr, nullptr, nullptr,
      Dw1, Db1, Dw2, Db2, Dw3, Db3, 0, (void*)h0h);
  // Layer 1: pre1 = h0*W1^T + b1 + x*Ws02^T + bs02
  fused_vecs_dan_kernel<true><<<grid, block, 0, stream>>>(
      h0h, W1, b1, xh, Ws02, bs02,
      Dw1, Db1, Dw2, Db2, Dw3, Db3, 1, (void*)h1h);
  // Layer 2: pre2 = h1*W2^T + b2 + h0*Ws13^T + bs13   (skip13 fused, never stored)
  fused_vecs_dan_kernel<false><<<grid, block, 0, stream>>>(
      h1h, W2, b2, h0h, Ws13, bs13,
      Dw1, Db1, Dw2, Db2, Dw3, Db3, 2, d_out);
}
